// WelschLoss_4724464026090
// MI455X (gfx1250) — compile-verified, matrix operand order checked by reading, not run on hardware
//
#include <hip/hip_runtime.h>
#include <hip/hip_bf16.h>
#include <math.h>

// ---------------- problem constants (from reference) ----------------
#define B_    8
#define N_    2048
#define M_    2048
#define D_    256
#define TOPK_ 128

constexpr float SK_EPS   = 0.001f;
constexpr float INV_EPS  = 1000.0f;
constexpr int   SK_ITERS = 20;
constexpr float CLIP_MIN = 0.0001f;
constexpr float ALPHA2   = 1.0f;

// ---------------- workspace layout (float units) ----------------
constexpr size_t SZ_COST  = (size_t)B_ * N_ * M_;           // 134 MB: fits in 192MB L2
constexpr size_t OFF_COST = 0;
constexpr size_t OFF_XN   = OFF_COST + SZ_COST;             // src feat sq-norms [B,N]
constexpr size_t OFF_YN   = OFF_XN  + (size_t)B_ * N_;      // tgt feat sq-norms [B,M]
constexpr size_t OFF_LOGP = OFF_YN  + (size_t)B_ * M_;
constexpr size_t OFF_LOGQ = OFF_LOGP + (size_t)B_ * N_;
constexpr size_t OFF_U    = OFF_LOGQ + (size_t)B_ * M_;
constexpr size_t OFF_V    = OFF_U   + (size_t)B_ * N_;
constexpr size_t OFF_CORR = OFF_V   + (size_t)B_ * M_;      // src_corr [B,N,3]
constexpr size_t OFF_PROB = OFF_CORR + (size_t)B_ * N_ * 3;
constexpr size_t OFF_SRCK = OFF_PROB + (size_t)B_ * N_;     // gathered src [B,K,3]
constexpr size_t OFF_TGTK = OFF_SRCK + (size_t)B_ * TOPK_ * 3;
constexpr size_t OFF_PART = OFF_TGTK + (size_t)B_ * TOPK_ * 3;
constexpr size_t OFF_IDX  = OFF_PART + B_;                  // int[B,K] stored in float slots

// ---------------- vector types for WMMA ----------------
typedef __attribute__((ext_vector_type(2)))  float  v2f;
typedef __attribute__((ext_vector_type(8)))  float  v8f;
typedef __attribute__((ext_vector_type(16))) __bf16 v16bf;

#if __has_builtin(__builtin_amdgcn_wmma_f32_16x16x4_f32)
#define HAVE_WMMA_F32 1
#endif

// ============================================================
// 1) per-row squared norms of src_feats / tgt_feats
//    one 256-thread block per row (D_ == 256)
// ============================================================
__global__ __launch_bounds__(256) void norms_kernel(const float* __restrict__ sf,
                                                    const float* __restrict__ tf,
                                                    float* __restrict__ ws) {
    __shared__ float red[256];
    int row = blockIdx.x;
    const float* p;
    float* out;
    if (row < B_ * N_) { p = sf + (size_t)row * D_;             out = ws + OFF_XN + row; }
    else               { int r = row - B_ * N_;
                         p = tf + (size_t)r * D_;               out = ws + OFF_YN + r; }
    float x = p[threadIdx.x];
    red[threadIdx.x] = x * x;
    __syncthreads();
    for (int s = 128; s > 0; s >>= 1) {
        if (threadIdx.x < s) red[threadIdx.x] += red[threadIdx.x + s];
        __syncthreads();
    }
    if (threadIdx.x == 0) *out = red[0];
}

// ============================================================
// 2) marginals: logp/logq = log(o) - log(max(sum(o), CLIP)); init u=v=0
// ============================================================
__global__ __launch_bounds__(256) void marg_kernel(const float* __restrict__ so,
                                                   const float* __restrict__ to,
                                                   float* __restrict__ ws) {
    __shared__ float red[256];
    __shared__ float lsum;
    int id = blockIdx.x;
    bool isSrc = id < B_;
    int b = isSrc ? id : id - B_;
    const float* o = isSrc ? (so + (size_t)b * N_) : (to + (size_t)b * M_);
    int n = isSrc ? N_ : M_;
    float s = 0.f;
    for (int i = threadIdx.x; i < n; i += 256) s += o[i];
    red[threadIdx.x] = s;
    __syncthreads();
    for (int k = 128; k > 0; k >>= 1) {
        if (threadIdx.x < k) red[threadIdx.x] += red[threadIdx.x + k];
        __syncthreads();
    }
    if (threadIdx.x == 0) lsum = __logf(fmaxf(red[0], CLIP_MIN));
    __syncthreads();
    float ls = lsum;
    float* lg = ws + (isSrc ? (OFF_LOGP + (size_t)b * N_) : (OFF_LOGQ + (size_t)b * M_));
    float* uv = ws + (isSrc ? (OFF_U    + (size_t)b * N_) : (OFF_V    + (size_t)b * M_));
    for (int i = threadIdx.x; i < n; i += 256) {
        lg[i] = __logf(o[i]) - ls;
        uv[i] = 0.f;
    }
}

// ============================================================
// 3) cost = cdist(src_feats, tgt_feats) via f32 WMMA (16x16x4)
//    block = (32,8) = 8 waves; each wave owns one 16x16 tile,
//    two independent WMMA accumulator chains over K to hide latency.
// ============================================================
__global__ __launch_bounds__(256) void cost_kernel(const float* __restrict__ sf,
                                                   const float* __restrict__ tf,
                                                   float* __restrict__ ws) {
    float* cost = ws + OFF_COST;
    const float* xn = ws + OFF_XN;
    const float* yn = ws + OFF_YN;

    int b    = blockIdx.z;
    int lane = threadIdx.x;        // 0..31 (wave32)
    int wv   = threadIdx.y;        // 0..7
    int row0 = blockIdx.y * 16;                 // M-tile (rows of cost)
    int col0 = (blockIdx.x * 8 + wv) * 16;      // N-tile (cols of cost)
    int l15  = lane & 15;
    int hi   = lane >> 4;

    const float* Ap = sf + ((size_t)b * N_ + row0 + l15) * D_;
    const float* Bp = tf + ((size_t)b * M_ + col0 + l15) * D_;

    v8f acc0 = {};
    v8f acc1 = {};
#ifdef HAVE_WMMA_F32
    // A 16x4 f32: lanes 0-15 hold K={0,1}, lanes 16-31 K={2,3} (VGPR0/1 per lane)
    int kh = hi * 2;
    for (int k = 0; k < D_; k += 8) {
        v2f a0; a0.x = Ap[k + kh];     a0.y = Ap[k + kh + 1];
        v2f b0; b0.x = Bp[k + kh];     b0.y = Bp[k + kh + 1];
        acc0 = __builtin_amdgcn_wmma_f32_16x16x4_f32(false, a0, false, b0,
                                                     (short)0, acc0, false, false);
        v2f a1; a1.x = Ap[k + 4 + kh]; a1.y = Ap[k + 4 + kh + 1];
        v2f b1; b1.x = Bp[k + 4 + kh]; b1.y = Bp[k + 4 + kh + 1];
        acc1 = __builtin_amdgcn_wmma_f32_16x16x4_f32(false, a1, false, b1,
                                                     (short)0, acc1, false, false);
    }
#else
    // fallback: codegen-confirmed bf16 WMMA path (16x16x32)
    for (int k = 0; k < D_; k += 64) {
        v16bf a0, b0, a1, b1;
        #pragma unroll
        for (int i = 0; i < 16; i++) {
            a0[i] = (__bf16)Ap[k + hi * 16 + i];
            b0[i] = (__bf16)Bp[k + hi * 16 + i];
            a1[i] = (__bf16)Ap[k + 32 + hi * 16 + i];
            b1[i] = (__bf16)Bp[k + 32 + hi * 16 + i];
        }
        acc0 = __builtin_amdgcn_wmma_f32_16x16x32_bf16(false, a0, false, b0,
                                                       (short)0, acc0, false, false);
        acc1 = __builtin_amdgcn_wmma_f32_16x16x32_bf16(false, a1, false, b1,
                                                       (short)0, acc1, false, false);
    }
#endif
    v8f dot = acc0 + acc1;

    // C/D layout: lane<16 -> N=lane, M=r; lane>=16 -> N=lane-16, M=r+8
    int ncol = col0 + l15;
    float yv = yn[(size_t)b * M_ + ncol];
    #pragma unroll
    for (int r = 0; r < 8; r++) {
        int mrow = row0 + r + 8 * hi;
        float d2 = xn[(size_t)b * N_ + mrow] + yv - 2.0f * dot[r];
        d2 = fmaxf(d2, 0.0f);
        cost[((size_t)b * N_ + mrow) * M_ + ncol] = sqrtf(d2);
    }
}

// ---------------- (max,sum) LSE combine helper ----------------
__device__ __forceinline__ void lse_merge(float& mx, float& s, float m2, float s2) {
    float mc = fmaxf(mx, m2);
    s = s * __expf(mx - mc) + s2 * __expf(m2 - mc);
    mx = mc;
}

// ============================================================
// 4) Sinkhorn row pass: u = eps*logp - eps*LSE_m((v - cost)/eps)
//    one block per (b,n), streaming row from L2
// ============================================================
__global__ __launch_bounds__(256) void row_lse_kernel(float* __restrict__ ws) {
    __shared__ float smx[256], ssm[256];
    int b = blockIdx.y, n = blockIdx.x, tid = threadIdx.x;
    const float* crow = ws + OFF_COST + ((size_t)b * N_ + n) * M_;
    const float* v    = ws + OFF_V + (size_t)b * M_;
    float mx = -3.4e38f, s = 0.f;
    for (int m = tid; m < M_; m += 256) {
        if (m + 256 < M_) __builtin_prefetch(crow + m + 256, 0, 1);
        float t = (v[m] - crow[m]) * INV_EPS;
        if (t > mx) { s = s * __expf(mx - t) + 1.f; mx = t; }
        else        { s += __expf(t - mx); }
    }
    smx[tid] = mx; ssm[tid] = s;
    __syncthreads();
    for (int k = 128; k > 0; k >>= 1) {
        if (tid < k) {
            float m1 = smx[tid], s1 = ssm[tid];
            lse_merge(m1, s1, smx[tid + k], ssm[tid + k]);
            smx[tid] = m1; ssm[tid] = s1;
        }
        __syncthreads();
    }
    if (tid == 0) {
        float L = smx[0] + __logf(ssm[0]);
        ws[OFF_U + (size_t)b * N_ + n] =
            SK_EPS * ws[OFF_LOGP + (size_t)b * N_ + n] - SK_EPS * L;
    }
}

// ============================================================
// 5) Sinkhorn column pass: v = eps*logq - eps*LSE_n((u - cost)/eps)
//    lane-per-column layout -> 128B coalesced column reads
// ============================================================
__global__ __launch_bounds__(256) void col_lse_kernel(float* __restrict__ ws) {
    __shared__ float smx[8][32], ssm[8][32];
    int b = blockIdx.y;
    int lane = threadIdx.x & 31;
    int wv   = threadIdx.x >> 5;
    int col  = blockIdx.x * 32 + lane;
    const float* cost = ws + OFF_COST + (size_t)b * N_ * M_;
    const float* u    = ws + OFF_U + (size_t)b * N_;
    float mx = -3.4e38f, s = 0.f;
    for (int n = wv; n < N_; n += 8) {
        float t = (u[n] - cost[(size_t)n * M_ + col]) * INV_EPS;
        if (t > mx) { s = s * __expf(mx - t) + 1.f; mx = t; }
        else        { s += __expf(t - mx); }
    }
    smx[wv][lane] = mx; ssm[wv][lane] = s;
    __syncthreads();
    if (wv == 0) {
        float m1 = smx[0][lane], s1 = ssm[0][lane];
        #pragma unroll
        for (int w = 1; w < 8; w++) lse_merge(m1, s1, smx[w][lane], ssm[w][lane]);
        float L = m1 + __logf(s1);
        ws[OFF_V + (size_t)b * M_ + col] =
            SK_EPS * ws[OFF_LOGQ + (size_t)b * M_ + col] - SK_EPS * L;
    }
}

// ============================================================
// 6) fused gamma pass: rowsum, gamma@tgt, rowmax(prob) in one read
// ============================================================
__global__ __launch_bounds__(256) void gamma_kernel(const float* __restrict__ tgt,
                                                    float* __restrict__ ws) {
    __shared__ float rs[256], r0[256], r1[256], r2[256], rp[256];
    int b = blockIdx.y, n = blockIdx.x, tid = threadIdx.x;
    const float* crow = ws + OFF_COST + ((size_t)b * N_ + n) * M_;
    const float* v    = ws + OFF_V + (size_t)b * M_;
    const float* tg   = tgt + (size_t)b * M_ * 3;
    float un = ws[OFF_U + (size_t)b * N_ + n];
    float s = 0.f, c0 = 0.f, c1 = 0.f, c2 = 0.f, p = 0.f;
    for (int m = tid; m < M_; m += 256) {
        float g = __expf((un + v[m] - crow[m]) * INV_EPS);
        s += g;
        p = fmaxf(p, g);
        c0 += g * tg[m * 3 + 0];
        c1 += g * tg[m * 3 + 1];
        c2 += g * tg[m * 3 + 2];
    }
    rs[tid] = s; r0[tid] = c0; r1[tid] = c1; r2[tid] = c2; rp[tid] = p;
    __syncthreads();
    for (int k = 128; k > 0; k >>= 1) {
        if (tid < k) {
            rs[tid] += rs[tid + k]; r0[tid] += r0[tid + k];
            r1[tid] += r1[tid + k]; r2[tid] += r2[tid + k];
            rp[tid] = fmaxf(rp[tid], rp[tid + k]);
        }
        __syncthreads();
    }
    if (tid == 0) {
        float inv = 1.f / fmaxf(rs[0], CLIP_MIN);
        float* corr = ws + OFF_CORR + ((size_t)b * N_ + n) * 3;
        corr[0] = r0[0] * inv; corr[1] = r1[0] * inv; corr[2] = r2[0] * inv;
        ws[OFF_PROB + (size_t)b * N_ + n] = rp[0];
    }
}

// ============================================================
// 7) top-K selection (K=128 of 2048) + gather src_k / tgt_k
//    deterministic: ties broken toward smallest index
// ============================================================
__global__ __launch_bounds__(256) void topk_kernel(const float* __restrict__ src,
                                                   float* __restrict__ ws) {
    __shared__ float bv[256];
    __shared__ int   bi[256];
    __shared__ unsigned int mask[N_ / 32];
    int b = blockIdx.x, tid = threadIdx.x;
    const float* prob = ws + OFF_PROB + (size_t)b * N_;
    int* idx = (int*)(ws + OFF_IDX) + (size_t)b * TOPK_;
    if (tid < N_ / 32) mask[tid] = 0u;
    __syncthreads();
    for (int k = 0; k < TOPK_; k++) {
        float best = -3.4e38f;
        int   bidx = 0x7fffffff;
        for (int i = tid; i < N_; i += 256) {
            if (!((mask[i >> 5] >> (i & 31)) & 1u)) {
                float pv = prob[i];
                if (pv > best || (pv == best && i < bidx)) { best = pv; bidx = i; }
            }
        }
        bv[tid] = best; bi[tid] = bidx;
        __syncthreads();
        for (int sr = 128; sr > 0; sr >>= 1) {
            if (tid < sr) {
                if (bv[tid + sr] > bv[tid] ||
                    (bv[tid + sr] == bv[tid] && bi[tid + sr] < bi[tid])) {
                    bv[tid] = bv[tid + sr]; bi[tid] = bi[tid + sr];
                }
            }
            __syncthreads();
        }
        if (tid == 0) {
            int w = bi[0];
            idx[k] = w;
            mask[w >> 5] |= (1u << (w & 31));
            const float* sp = src + ((size_t)b * N_ + w) * 3;
            const float* cp = ws + OFF_CORR + ((size_t)b * N_ + w) * 3;
            float* sk = ws + OFF_SRCK + ((size_t)b * TOPK_ + k) * 3;
            float* tk = ws + OFF_TGTK + ((size_t)b * TOPK_ + k) * 3;
            sk[0] = sp[0]; sk[1] = sp[1]; sk[2] = sp[2];
            tk[0] = cp[0]; tk[1] = cp[1]; tk[2] = cp[2];
        }
        __syncthreads();
    }
}

// ============================================================
// 8) K x K Welsch edge loss per batch -> partial[b]
// ============================================================
__global__ __launch_bounds__(256) void loss_kernel(float* __restrict__ ws) {
    __shared__ float red[256];
    int b = blockIdx.x, tid = threadIdx.x;
    const float* sk = ws + OFF_SRCK + (size_t)b * TOPK_ * 3;
    const float* tk = ws + OFF_TGTK + (size_t)b * TOPK_ * 3;
    float acc = 0.f;
    for (int p = tid; p < TOPK_ * TOPK_; p += 256) {
        int i = p >> 7, j = p & (TOPK_ - 1);
        float dx = sk[i * 3 + 0] - sk[j * 3 + 0];
        float dy = sk[i * 3 + 1] - sk[j * 3 + 1];
        float dz = sk[i * 3 + 2] - sk[j * 3 + 2];
        float se = sqrtf(fmaxf(dx * dx + dy * dy + dz * dz, 0.f));
        float ex = tk[i * 3 + 0] - tk[j * 3 + 0];
        float ey = tk[i * 3 + 1] - tk[j * 3 + 1];
        float ez = tk[i * 3 + 2] - tk[j * 3 + 2];
        float te = sqrtf(fmaxf(ex * ex + ey * ey + ez * ez, 0.f));
        float z = fabsf(se - te);
        acc += 1.f - __expf(-0.5f * z * z / ALPHA2);
    }
    red[tid] = acc;
    __syncthreads();
    for (int k = 128; k > 0; k >>= 1) {
        if (tid < k) red[tid] += red[tid + k];
        __syncthreads();
    }
    if (tid == 0) ws[OFF_PART + b] = red[0];
}

// ============================================================
// 9) final scalar: mean over [B, K]
// ============================================================
__global__ void final_kernel(const float* __restrict__ ws, float* __restrict__ out) {
    if (threadIdx.x == 0) {
        float s = 0.f;
        for (int b = 0; b < B_; b++) s += ws[OFF_PART + b];
        out[0] = s / (float)(B_ * TOPK_);
    }
}

// ============================================================
// host launcher
// ============================================================
extern "C" void kernel_launch(void* const* d_in, const int* in_sizes, int n_in,
                              void* d_out, int out_size, void* d_ws, size_t ws_size,
                              hipStream_t stream) {
    const float* src = (const float*)d_in[0];   // [B,N,3]
    const float* tgt = (const float*)d_in[1];   // [B,M,3]
    const float* sf  = (const float*)d_in[2];   // [B,N,D]
    const float* tf  = (const float*)d_in[3];   // [B,M,D]
    const float* so  = (const float*)d_in[4];   // [B,N]
    const float* to  = (const float*)d_in[5];   // [B,M]
    float* ws  = (float*)d_ws;
    float* out = (float*)d_out;

    norms_kernel<<<B_ * (N_ + M_), 256, 0, stream>>>(sf, tf, ws);
    marg_kernel<<<2 * B_, 256, 0, stream>>>(so, to, ws);

    dim3 cg(M_ / 128, N_ / 16, B_);
    dim3 cb(32, 8);
    cost_kernel<<<cg, cb, 0, stream>>>(sf, tf, ws);

    for (int it = 0; it < SK_ITERS; ++it) {
        row_lse_kernel<<<dim3(N_, B_), 256, 0, stream>>>(ws);
        col_lse_kernel<<<dim3(M_ / 32, B_), 256, 0, stream>>>(ws);
    }

    gamma_kernel<<<dim3(N_, B_), 256, 0, stream>>>(tgt, ws);
    topk_kernel<<<B_, 256, 0, stream>>>(src, ws);
    loss_kernel<<<B_, 256, 0, stream>>>(ws);
    final_kernel<<<1, 32, 0, stream>>>(ws, out);
}
